// FourierUnit_29600914604301
// MI455X (gfx1250) — matrix operand order repaired and checked by reading, not compiled, not run on hardware
//
#include <hip/hip_runtime.h>
#include <hip/hip_bf16.h>
#include <math.h>
#include <stdint.h>

// ---------------------------------------------------------------------------
// FourierUnit on MI455X (gfx1250, wave32, WMMA).
//
// Split-bf16 storage (hi + residual-lo planes, same bytes as fp32) for every
// GEMM operand; 3 WMMA products (hh + hl + lh) give ~2^-17 precision with
// zero conversion VALU in the hot loop.  GEMM hot loop = double-buffered
// async global->LDS DMA (ASYNCcnt) overlapped with ds_load_b128 fragment
// reads + v_wmma_f32_16x16x32_bf16.
//
// Stages per half: W-DFT -> H-DFT -> 1x1 conv -> GroupNorm+ReLU ->
// inverse H-DFT (only needed 128 rows) -> Hermitian-folded inverse W-DFT
// written straight into d_out.  Scales/signs baked into generated bases.
// ---------------------------------------------------------------------------

typedef __attribute__((ext_vector_type(16))) __bf16 v16bf;
typedef __attribute__((ext_vector_type(8)))  __bf16 v8bf;
typedef __attribute__((ext_vector_type(8)))  float  v8f;

#define BM 128
#define BK 32
#define KP 40            // padded K stride (bf16): 80B rows, 16B aligned

// ---- CDNA5 async global->LDS copy (ASYNCcnt path) -------------------------
__device__ __forceinline__ void async_copy_b128(unsigned lds_byte_addr, const void* g) {
  asm volatile("global_load_async_to_lds_b128 %0, %1, off"
               :: "v"(lds_byte_addr), "v"((unsigned long long)(uintptr_t)g)
               : "memory");
}
__device__ __forceinline__ void wait_async0() {
  asm volatile("s_wait_asynccnt 0" ::: "memory");
}

struct GemmArgs {
  const __bf16 *A1h, *A1l, *A2h, *A2l;  // row-major [m][k], ld=ldA (A2h==null => 1 product)
  const __bf16 *B1h, *B1l, *B2h, *B2l;  // transposed [n][k], ld=ldB
  __bf16 *Dh, *Dl;                      // split output (if Dl != nullptr)
  float  *Df;                           // fp32 output (if Dl == nullptr)
  const float* bias;                    // indexed by m, may be null
  int M, N, K, nc;                      // z = b*nc + c
  long long ldA, ldB;
  long long sDm, sDn;                   // D addr = off + m*sDm + n*sDn
  long long sAc, sAb, sBc, sBb, sDc, sDb;
};

// BNT = 48 (6 waves) for N=129 stages, 64 (8 waves) for N=256/32895 stages.
template<int BNT>
__global__ __launch_bounds__(2 * (BNT / 16) * 32) void wmma_gemm(GemmArgs g) {
  constexpr int NTH = 2 * (BNT / 16) * 32;
  __shared__ __align__(16) __bf16 Ahs[2][BM][KP];
  __shared__ __align__(16) __bf16 Als[2][BM][KP];
  __shared__ __align__(16) __bf16 Bhs[2][BNT][KP];
  __shared__ __align__(16) __bf16 Bls[2][BNT][KP];

  const int t    = threadIdx.x;
  const int lane = t & 31;
  const int wave = t >> 5;
  const int wm   = wave & 1;    // M half (64 rows)
  const int wn   = wave >> 1;   // 0..BNT/16-1, 16 cols each
  const int ln   = lane & 15;
  const int lh   = lane >> 4;

  const int m0 = blockIdx.y * BM;
  const int n0 = blockIdx.x * BNT;
  const int bo = blockIdx.z / g.nc;
  const int bc = blockIdx.z % g.nc;

  const long long offA = (long long)bo * g.sAb + (long long)bc * g.sAc
                       + (long long)m0 * g.ldA;
  const long long offB = (long long)bo * g.sBb + (long long)bc * g.sBc
                       + (long long)n0 * g.ldB;
  const long long offD = (long long)bo * g.sDb + (long long)bc * g.sDc;

  const int mrem = g.M - m0;
  const int nrem = g.N - n0;

  v8f acc[4] = {};
  const int nprod  = (g.A2h != nullptr) ? 2 : 1;
  const int nk     = (g.K + BK - 1) / BK;
  const int nsteps = nprod * nk;

  // ---- stage one (product, k) tile pair into LDS buffer `buf` -------------
  auto stage = [&](int buf, int s) {
    const int p    = (s >= nk) ? 1 : 0;
    const int k0   = (p ? s - nk : s) * BK;
    const int krem = g.K - k0;
    const __bf16* Aph = (p ? g.A2h : g.A1h) + offA;
    const __bf16* Apl = (p ? g.A2l : g.A1l) + offA;
    const __bf16* Bph = (p ? g.B2h : g.B1h) + offB;
    const __bf16* Bpl = (p ? g.B2l : g.B1l) + offB;

    constexpr int ACH = (BM * BK) / 8;    // 8-elem chunks per plane
    constexpr int BCH = (BNT * BK) / 8;

    if (mrem >= BM && krem >= BK) {       // interior fast path (uniform)
      for (int slot = t; slot < 2 * ACH; slot += NTH) {
        int pl = slot >= ACH, rem = slot - pl * ACH;
        int r = rem >> 2, c8 = (rem & 3) << 3;
        async_copy_b128((unsigned)(uintptr_t)(pl ? &Als[buf][r][c8] : &Ahs[buf][r][c8]),
                        (pl ? Apl : Aph) + (long long)r * g.ldA + k0 + c8);
      }
    } else {
      for (int slot = t; slot < 2 * ACH; slot += NTH) {
        int pl = slot >= ACH, rem = slot - pl * ACH;
        int r = rem >> 2, c8 = (rem & 3) << 3;
        const __bf16* src = (pl ? Apl : Aph) + (long long)r * g.ldA + k0 + c8;
        __bf16* dst = pl ? &Als[buf][r][c8] : &Ahs[buf][r][c8];
        if (r < mrem && (c8 + 8) <= krem) {
          async_copy_b128((unsigned)(uintptr_t)dst, src);
        } else {
          #pragma unroll
          for (int j = 0; j < 8; ++j) {
            __bf16 v = (__bf16)0.f;
            if (r < mrem && (c8 + j) < krem) v = src[j];
            dst[j] = v;
          }
        }
      }
    }
    if (nrem >= BNT && krem >= BK) {
      for (int slot = t; slot < 2 * BCH; slot += NTH) {
        int pl = slot >= BCH, rem = slot - pl * BCH;
        int r = rem >> 2, c8 = (rem & 3) << 3;
        async_copy_b128((unsigned)(uintptr_t)(pl ? &Bls[buf][r][c8] : &Bhs[buf][r][c8]),
                        (pl ? Bpl : Bph) + (long long)r * g.ldB + k0 + c8);
      }
    } else {
      for (int slot = t; slot < 2 * BCH; slot += NTH) {
        int pl = slot >= BCH, rem = slot - pl * BCH;
        int r = rem >> 2, c8 = (rem & 3) << 3;
        const __bf16* src = (pl ? Bpl : Bph) + (long long)r * g.ldB + k0 + c8;
        __bf16* dst = pl ? &Bls[buf][r][c8] : &Bhs[buf][r][c8];
        if (r < nrem && (c8 + 8) <= krem) {
          async_copy_b128((unsigned)(uintptr_t)dst, src);
        } else {
          #pragma unroll
          for (int j = 0; j < 8; ++j) {
            __bf16 v = (__bf16)0.f;
            if (r < nrem && (c8 + j) < krem) v = src[j];
            dst[j] = v;
          }
        }
      }
    }
  };

  // ---- compute one staged tile pair ---------------------------------------
  auto compute = [&](int buf) {
    const int nrow = wn * 16 + ln;
    const int kb2  = lh * 16;
    v8bf b0h = *(const v8bf*)(&Bhs[buf][nrow][kb2]);
    v8bf b1h = *(const v8bf*)(&Bhs[buf][nrow][kb2 + 8]);
    v8bf b0l = *(const v8bf*)(&Bls[buf][nrow][kb2]);
    v8bf b1l = *(const v8bf*)(&Bls[buf][nrow][kb2 + 8]);
    v16bf bh = __builtin_shufflevector(b0h, b1h, 0,1,2,3,4,5,6,7,8,9,10,11,12,13,14,15);
    v16bf bl = __builtin_shufflevector(b0l, b1l, 0,1,2,3,4,5,6,7,8,9,10,11,12,13,14,15);

    const int kb = lh * 16;  // lanes 0-15: K {0..15}; 16-31: {16..31}
    #pragma unroll
    for (int sub = 0; sub < 4; ++sub) {
      const int mrow = wm * 64 + sub * 16 + ln;
      v8bf a0h = *(const v8bf*)(&Ahs[buf][mrow][kb]);
      v8bf a1h = *(const v8bf*)(&Ahs[buf][mrow][kb + 8]);
      v8bf a0l = *(const v8bf*)(&Als[buf][mrow][kb]);
      v8bf a1l = *(const v8bf*)(&Als[buf][mrow][kb + 8]);
      v16bf ah = __builtin_shufflevector(a0h, a1h, 0,1,2,3,4,5,6,7,8,9,10,11,12,13,14,15);
      v16bf al = __builtin_shufflevector(a0l, a1l, 0,1,2,3,4,5,6,7,8,9,10,11,12,13,14,15);
      acc[sub] = __builtin_amdgcn_wmma_f32_16x16x32_bf16(
          false, ah, false, bh, (short)0, acc[sub], false, false);
      acc[sub] = __builtin_amdgcn_wmma_f32_16x16x32_bf16(
          false, ah, false, bl, (short)0, acc[sub], false, false);
      acc[sub] = __builtin_amdgcn_wmma_f32_16x16x32_bf16(
          false, al, false, bh, (short)0, acc[sub], false, false);
    }
  };

  // ---- software pipeline: DMA(step s+1) overlaps WMMA(step s) -------------
  stage(0, 0);
  for (int s = 0; s < nsteps; ++s) {
    wait_async0();
    __syncthreads();
    if (s + 1 < nsteps) stage((s + 1) & 1, s + 1);
    compute(s & 1);
  }

  // ---- epilogue -----------------------------------------------------------
  const int ncol = n0 + wn * 16 + ln;
  if (ncol < g.N) {
    #pragma unroll
    for (int sub = 0; sub < 4; ++sub) {
      const int mb = m0 + wm * 64 + sub * 16 + lh * 8;
      #pragma unroll
      for (int r = 0; r < 8; ++r) {
        const int m = mb + r;
        if (m < g.M) {
          float v = acc[sub][r];
          if (g.bias) v += g.bias[m];
          const long long a = offD + (long long)m * g.sDm + (long long)ncol * g.sDn;
          if (g.Dl) {
            __bf16 h = (__bf16)v;
            g.Dh[a] = h;
            g.Dl[a] = (__bf16)(v - (float)h);
          } else {
            g.Df[a] = v;
          }
        }
      }
    }
  }
}

// ---------------------------------------------------------------------------
// Basis generation, directly split into bf16 hi/lo planes.
// v = sign * scale * w * trig(2*pi*r*c/Ndft); trig: 0=cos 1=sin;
// hermcol: w = (c==0 || c==Ndft/2) ? 1 : 2.
// ---------------------------------------------------------------------------
__global__ void dft_init_split(__bf16* hi, __bf16* lo, int rows, int cols,
                               int Ndft, int trig, float sign, float scale,
                               int hermcol) {
  long long idx = (long long)blockIdx.x * blockDim.x + threadIdx.x;
  if (idx >= (long long)rows * cols) return;
  int r = (int)(idx / cols), c = (int)(idx % cols);
  double ang = 2.0 * 3.14159265358979323846 * (double)r * (double)c / (double)Ndft;
  double tv  = trig ? sin(ang) : cos(ang);
  float w = 1.f;
  if (hermcol) w = (c == 0 || c == Ndft / 2) ? 1.f : 2.f;
  float v = sign * scale * w * (float)tv;
  __bf16 h = (__bf16)v;
  hi[idx] = h;
  lo[idx] = (__bf16)(v - (float)h);
}

__global__ void split_f32(const float* in, __bf16* hi, __bf16* lo, long long n) {
  long long idx = (long long)blockIdx.x * blockDim.x + threadIdx.x;
  if (idx >= n) return;
  float v = in[idx];
  __bf16 h = (__bf16)v;
  hi[idx] = h;
  lo[idx] = (__bf16)(v - (float)h);
}

// ---------------------------------------------------------------------------
// GroupNorm(16) + ReLU, deterministic 3-kernel scheme (no float atomics).
// 64 groups of 8*S elements; CHUNKS partial blocks per group.
// ---------------------------------------------------------------------------
#define GN_CHUNKS 32

__global__ __launch_bounds__(256) void gn_partial(
    const __bf16* Uh, const __bf16* Ul, float* part, int S) {
  const int grp = blockIdx.x >> 5;           // 0..63  (b*16 + g)
  const int ck  = blockIdx.x & 31;
  const long long len   = 8LL * S;
  const long long base  = (long long)grp * len;   // groups are contiguous
  const long long start = ck * len / GN_CHUNKS;
  const long long end   = (ck + 1) * len / GN_CHUNKS;

  float s = 0.f, ss = 0.f;
  for (long long i = start + threadIdx.x; i < end; i += 256) {
    float x = (float)Uh[base + i] + (float)Ul[base + i];
    s += x; ss += x * x;
  }
  __shared__ float rs[256], rq[256];
  rs[threadIdx.x] = s; rq[threadIdx.x] = ss;
  __syncthreads();
  for (int o = 128; o > 0; o >>= 1) {
    if ((int)threadIdx.x < o) {
      rs[threadIdx.x] += rs[threadIdx.x + o];
      rq[threadIdx.x] += rq[threadIdx.x + o];
    }
    __syncthreads();
  }
  if (threadIdx.x == 0) {
    part[blockIdx.x] = rs[0];
    part[2048 + blockIdx.x] = rq[0];
  }
}

__global__ void gn_reduce(const float* part, float* stats, int S) {
  const int grp = threadIdx.x;  // 64 threads
  if (grp >= 64) return;
  float s = 0.f, ss = 0.f;
  for (int c = 0; c < GN_CHUNKS; ++c) {
    s  += part[grp * GN_CHUNKS + c];
    ss += part[2048 + grp * GN_CHUNKS + c];
  }
  const float n  = 8.f * (float)S;
  const float mu = s / n;
  const float vr = ss / n - mu * mu;
  stats[grp]      = mu;
  stats[64 + grp] = rsqrtf(vr + 1e-5f);
}

__global__ __launch_bounds__(256) void gn_apply(
    __bf16* Uh, __bf16* Ul, const float* stats,
    const float* gamma, const float* beta, int S) {
  long long idx = (long long)blockIdx.x * blockDim.x + threadIdx.x;
  const long long total = 4LL * 128 * S;
  if (idx >= total) return;
  int ch  = (int)((idx / S) & 127);
  int grp = (int)(idx / (8LL * S));     // (b*16 + g), groups contiguous
  float x = (float)Uh[idx] + (float)Ul[idx];
  x = (x - stats[grp]) * stats[64 + grp] * gamma[ch] + beta[ch];
  x = x > 0.f ? x : 0.f;
  __bf16 h = (__bf16)x;
  Uh[idx] = h;
  Ul[idx] = (__bf16)(x - (float)h);
}

// ---------------------------------------------------------------------------
template<int BNT>
static inline void gemm(hipStream_t st,
                        const __bf16* A1h, const __bf16* A1l,
                        const __bf16* A2h, const __bf16* A2l,
                        const __bf16* B1h, const __bf16* B1l,
                        const __bf16* B2h, const __bf16* B2l,
                        __bf16* Dh, __bf16* Dl, float* Df, const float* bias,
                        int M, int N, int K, int nc, int nb,
                        long long ldA, long long ldB,
                        long long sDm, long long sDn,
                        long long sAc, long long sAb,
                        long long sBc, long long sBb,
                        long long sDc, long long sDb) {
  GemmArgs g{A1h, A1l, A2h, A2l, B1h, B1l, B2h, B2l, Dh, Dl, Df, bias,
             M, N, K, nc, ldA, ldB, sDm, sDn, sAc, sAb, sBc, sBb, sDc, sDb};
  dim3 grid((unsigned)((N + BNT - 1) / BNT), (unsigned)((M + BM - 1) / BM), (unsigned)nb);
  hipLaunchKernelGGL(wmma_gemm<BNT>, grid, dim3(2 * (BNT / 16) * 32), 0, st, g);
}

extern "C" void kernel_launch(void* const* d_in, const int* in_sizes, int n_in,
                              void* d_out, int out_size, void* d_ws, size_t ws_size,
                              hipStream_t stream) {
  (void)in_sizes; (void)n_in; (void)out_size; (void)ws_size;

  const float* x      = (const float*)d_in[0];   // (4,64,256,256)
  const float* conv_w = (const float*)d_in[1];   // (128,128) [o,c]
  const float* conv_b = (const float*)d_in[2];
  const float* gamma  = (const float*)d_in[3];
  const float* beta   = (const float*)d_in[4];
  float* out = (float*)d_out;                    // (4,64,256,256)

  const long long S  = 255LL * 129;   // 32895 (fw-major plane for Z/U)
  const long long SV = 128LL * 129;   // 16512
  const long long NX = 4LL * 64 * 256 * 256;

  // ---- workspace carve-up (bf16 elements, 128B aligned chunks) ----
  __bf16* wsb = (__bf16*)d_ws;
  long long cur = 0;
  auto take = [&](long long n) { __bf16* p = wsb + cur; cur += (n + 63) & ~63LL; return p; };

  __bf16 *CWFh = take(129 * 256), *CWFl = take(129 * 256);   // B-op [129][256]
  __bf16 *SWFh = take(129 * 256), *SWFl = take(129 * 256);
  __bf16 *CHh_ = take(255 * 255), *CHl_ = take(255 * 255);   // A-op [255][255]
  __bf16 *SHh_ = take(255 * 255), *SHl_ = take(255 * 255);
  __bf16 *SNh_ = take(255 * 255), *SNl_ = take(255 * 255);   // -sin
  __bf16 *CIh_ = take(255 * 255), *CIl_ = take(255 * 255);   // cos/255
  __bf16 *SIh_ = take(255 * 255), *SIl_ = take(255 * 255);   // sin/255
  __bf16 *NIh_ = take(255 * 255), *NIl_ = take(255 * 255);   // -sin/255
  __bf16 *ICh  = take(256 * 129), *ICl  = take(256 * 129);   // B-op [256][129]
  __bf16 *ISh  = take(256 * 129), *ISl  = take(256 * 129);
  __bf16 *Wh   = take(128 * 128), *Wl   = take(128 * 128);
  __bf16 *Xh   = take(NX),        *Xl   = take(NX);
  float  *gnws = (float*)take(2 * (4096 + 128) + 64);        // partials + stats
  __bf16* bufA = take(4LL * 256 * S);        // Y planes, later U planes
  __bf16* bufB = take(2LL * 4 * S * 128);    // Z planes, later V planes

  auto g1 = [](long long n) { return dim3((unsigned)((n + 255) / 256)); };
  dft_init_split<<<g1(129 * 256), 256, 0, stream>>>(CWFh, CWFl, 129, 256, 256, 0,  1.f, 1.f, 0);
  dft_init_split<<<g1(129 * 256), 256, 0, stream>>>(SWFh, SWFl, 129, 256, 256, 1, -1.f, 1.f, 0);
  dft_init_split<<<g1(255 * 255), 256, 0, stream>>>(CHh_, CHl_, 255, 255, 255, 0,  1.f, 1.f, 0);
  dft_init_split<<<g1(255 * 255), 256, 0, stream>>>(SHh_, SHl_, 255, 255, 255, 1,  1.f, 1.f, 0);
  dft_init_split<<<g1(255 * 255), 256, 0, stream>>>(SNh_, SNl_, 255, 255, 255, 1, -1.f, 1.f, 0);
  dft_init_split<<<g1(255 * 255), 256, 0, stream>>>(CIh_, CIl_, 255, 255, 255, 0,  1.f, 1.f / 255.f, 0);
  dft_init_split<<<g1(255 * 255), 256, 0, stream>>>(SIh_, SIl_, 255, 255, 255, 1,  1.f, 1.f / 255.f, 0);
  dft_init_split<<<g1(255 * 255), 256, 0, stream>>>(NIh_, NIl_, 255, 255, 255, 1, -1.f, 1.f / 255.f, 0);
  dft_init_split<<<g1(256 * 129), 256, 0, stream>>>(ICh, ICl, 256, 129, 256, 0,  1.f, 1.f / 256.f, 1);
  dft_init_split<<<g1(256 * 129), 256, 0, stream>>>(ISh, ISl, 256, 129, 256, 1, -1.f, 1.f / 256.f, 1);
  split_f32<<<g1(NX), 256, 0, stream>>>(x, Xh, Xl, NX);
  split_f32<<<g1(128 * 128), 256, 0, stream>>>(conv_w, Wh, Wl, 128 * 128);

  for (int half = 0; half < 2; ++half) {
    const long long xoff = half ? 256 : 0;            // row slice
    const long long hoff = half ? 127LL * 255 : 0;    // inverse-H row window
    float* Dout = out + (half ? 128LL * 256 : 0);

    __bf16 *Yrh = bufA,                 *Yrl = bufA + 256LL * S;   // [fw][h] per (b,c)
    __bf16 *Yih = bufA + 2 * 256LL * S, *Yil = bufA + 3 * 256LL * S;
    __bf16 *Zh = bufB, *Zl = bufB + 4LL * S * 128;                 // (b, p, ch)
    __bf16 *Uh = bufA, *Ul = bufA + 2 * 256LL * S;                 // (b, o, p)
    __bf16 *Vrh = bufB,                  *Vrl = bufB + 256LL * SV; // [h][fw] per (b,c)
    __bf16 *Vih = bufB + 2 * 256LL * SV, *Vil = bufB + 3 * 256LL * SV;

    // A: forward W-DFT. D -> Y^T (sDm=1, sDn=255)
    gemm<48>(stream, Xh + xoff, Xl + xoff, nullptr, nullptr,
             CWFh, CWFl, nullptr, nullptr, Yrh, Yrl, nullptr, nullptr,
             255, 129, 256, 64, 256,
             256, 256, 1, 255,
             65536, 64LL * 65536, 0, 0, S, 64LL * S);
    gemm<48>(stream, Xh + xoff, Xl + xoff, nullptr, nullptr,
             SWFh, SWFl, nullptr, nullptr, Yih, Yil, nullptr, nullptr,
             255, 129, 256, 64, 256,
             256, 256, 1, 255,
             65536, 64LL * 65536, 0, 0, S, 64LL * S);

    // B: forward H-DFT. D -> Z^T: addr = (n*255+m)*128 + c
    gemm<48>(stream, CHh_, CHl_, SHh_, SHl_,
             Yrh, Yrl, Yih, Yil, Zh, Zl, nullptr, nullptr,
             255, 129, 255, 64, 256,
             255, 255, 128, 255LL * 128,
             0, 0, S, 64LL * S, 1, 32895LL * 128);
    gemm<48>(stream, CHh_, CHl_, SNh_, SNl_,
             Yih, Yil, Yrh, Yrl, Zh + 64, Zl + 64, nullptr, nullptr,
             255, 129, 255, 64, 256,
             255, 255, 128, 255LL * 128,
             0, 0, S, 64LL * S, 1, 32895LL * 128);

    // C: 1x1 conv. B = Z^T [p][ch]; D -> U (b,o,p) row-major
    gemm<64>(stream, Wh, Wl, nullptr, nullptr,
             Zh, Zl, nullptr, nullptr, Uh, Ul, nullptr, conv_b,
             128, (int)S, 128, 1, 4,
             128, 128, S, 1,
             0, 0, 0, S * 128, 0, 128LL * S);

    // D: GroupNorm + ReLU (deterministic partial/reduce/apply)
    gn_partial<<<dim3(64 * GN_CHUNKS), 256, 0, stream>>>(Uh, Ul, gnws, (int)S);
    gn_reduce<<<dim3(1), 64, 0, stream>>>(gnws, gnws + 4096, (int)S);
    gn_apply<<<g1(4LL * 128 * S), 256, 0, stream>>>(Uh, Ul, gnws + 4096, gamma, beta, (int)S);

    // E: inverse H-DFT (128 rows). B = U viewed [fw][fh] (ld=255); D -> V
    gemm<48>(stream, CIh_ + hoff, CIl_ + hoff, NIh_ + hoff, NIl_ + hoff,
             Uh, Ul, Uh + 64LL * S, Ul + 64LL * S, Vrh, Vrl, nullptr, nullptr,
             128, 129, 255, 64, 256,
             255, 255, 129, 1,
             0, 0, S, 128LL * S, SV, 64LL * SV);
    gemm<48>(stream, CIh_ + hoff, CIl_ + hoff, SIh_ + hoff, SIl_ + hoff,
             Uh + 64LL * S, Ul + 64LL * S, Uh, Ul, Vih, Vil, nullptr, nullptr,
             128, 129, 255, 64, 256,
             255, 255, 129, 1,
             0, 0, S, 128LL * S, SV, 64LL * SV);

    // F: inverse W-DFT -> fp32 straight into d_out
    gemm<64>(stream, Vrh, Vrl, Vih, Vil,
             ICh, ICl, ISh, ISl, nullptr, nullptr, Dout, nullptr,
             128, 256, 129, 64, 256,
             129, 129, 256, 1,
             SV, 64LL * SV, 0, 0, 65536, 64LL * 65536);
  }
}
